// MultiHeadAttention_4346506903786
// MI455X (gfx1250) — matrix operand order repaired and checked by reading, not compiled, not run on hardware
//
#include <hip/hip_runtime.h>
#include <math.h>

// ---------------- problem constants ----------------
#define BB   2
#define SS   2048
#define DMOD 1024
#define HH   16
#define DD   64
#define EPSY 1e-5f

typedef __bf16 bf16;
typedef __attribute__((ext_vector_type(16))) __bf16 v16bf;
typedef __attribute__((ext_vector_type(8)))  __bf16 v8bf;
typedef __attribute__((ext_vector_type(8)))  float  v8f;

union FragBF {
    v16bf v;
    v8bf  h[2];
};

// ---------------------------------------------------------------------------
// Fragment loaders (bf16, wave32 WMMA 16x16x32 layouts per CDNA5 ISA 7.12.2)
// A (16x32): lane l -> row l%16 ; k-chunks: [half*8 .. +7] and [16+half*8 .. +7]
// B (32x16): lane l -> col l%16 ; k-chunks: [half*8 .. +7] and [16+half*8 .. +7]
// ---------------------------------------------------------------------------
static __device__ __forceinline__ v16bf load_frag_a(const bf16* __restrict__ base,
                                                    int row, int ldk, int kbase, int half) {
    FragBF f;
    const bf16* p = base + (size_t)row * ldk + kbase + half * 8;
    f.h[0] = *(const v8bf*)(p);
    f.h[1] = *(const v8bf*)(p + 16);
    return f.v;
}

static __device__ __forceinline__ v16bf load_frag_b(const bf16* __restrict__ baseT,
                                                    int col, int ldk, int kbase, int half) {
    FragBF f;
    const bf16* p = baseT + (size_t)col * ldk + kbase + half * 8;
    f.h[0] = *(const v8bf*)(p);
    f.h[1] = *(const v8bf*)(p + 16);
    return f.v;
}

// ---------------------------------------------------------------------------
// f32 -> bf16 elementwise convert
// ---------------------------------------------------------------------------
__global__ void cvt_f32_bf16_kernel(const float* __restrict__ src,
                                    bf16* __restrict__ dst, int n) {
    int i = blockIdx.x * blockDim.x + threadIdx.x;
    if (i < n) dst[i] = (bf16)src[i];
}

// ---------------------------------------------------------------------------
// weight transpose + convert: wt[n][k] = (bf16) w[k][n]   (DMOD x DMOD)
// ---------------------------------------------------------------------------
__global__ void transpose_cvt_kernel(const float* __restrict__ w,
                                     bf16* __restrict__ wt) {
    int i = blockIdx.x * blockDim.x + threadIdx.x;   // i = n*DMOD + k
    if (i >= DMOD * DMOD) return;
    int n = i >> 10;
    int k = i & (DMOD - 1);
    wt[i] = (bf16)w[(size_t)k * DMOD + n];
}

// ---------------------------------------------------------------------------
// Fused QKV projection GEMM (bf16 WMMA, f32 accum).
// Block = 256 threads = 8 waves as 2(M) x 4(N); wave tile 64x64 (4x4 WMMA
// tiles -> 16 b128 loads feed 16 WMMAs per k-step); WG tile 128x256.
// grid = (4096/128, 1024/256, 3)  z: 0=Q, 1=K, 2=V
// Q,K stored [B,H,S,D] bf16 ; V stored transposed [B,H,D,S] bf16.
// ---------------------------------------------------------------------------
__global__ void __launch_bounds__(256) qkv_gemm_kernel(
    const bf16* __restrict__ Xb,
    const bf16* __restrict__ WqT, const bf16* __restrict__ WkT, const bf16* __restrict__ WvT,
    const float* __restrict__ bq, const float* __restrict__ bk, const float* __restrict__ bv,
    bf16* __restrict__ Qo, bf16* __restrict__ Ko, bf16* __restrict__ Vt)
{
    const int wave = threadIdx.x >> 5;
    const int wm   = wave >> 2, wn = wave & 3;
    const int lane = threadIdx.x & 31;
    const int half = lane >> 4, lm = lane & 15;

    const int m0 = blockIdx.x * 128 + wm * 64;
    const int n0 = blockIdx.y * 256 + wn * 64;
    const int z  = blockIdx.z;

    const bf16*  WT   = (z == 0) ? WqT : (z == 1) ? WkT : WvT;
    const float* bias = (z == 0) ? bq  : (z == 1) ? bk  : bv;

    v8f acc[4][4];
#pragma unroll
    for (int mt = 0; mt < 4; mt++)
#pragma unroll
        for (int nt = 0; nt < 4; nt++)
#pragma unroll
            for (int e = 0; e < 8; e++) acc[mt][nt][e] = 0.0f;

    for (int kb = 0; kb < DMOD; kb += 32) {
        v16bf af[4], bfm[4];
#pragma unroll
        for (int mt = 0; mt < 4; mt++)
            af[mt] = load_frag_a(Xb, m0 + mt * 16 + lm, DMOD, kb, half);
#pragma unroll
        for (int nt = 0; nt < 4; nt++)
            bfm[nt] = load_frag_b(WT, n0 + nt * 16 + lm, DMOD, kb, half);
#pragma unroll
        for (int mt = 0; mt < 4; mt++)
#pragma unroll
            for (int nt = 0; nt < 4; nt++)
                acc[mt][nt] = __builtin_amdgcn_wmma_f32_16x16x32_bf16(
                    false, af[mt], false, bfm[nt], (short)0, acc[mt][nt], false, false);
    }

#pragma unroll
    for (int mt = 0; mt < 4; mt++) {
#pragma unroll
        for (int nt = 0; nt < 4; nt++) {
            const int n  = n0 + nt * 16 + lm;
            const float bn = bias[n];
            const int hh = n >> 6, dd = n & 63;
#pragma unroll
            for (int e = 0; e < 8; e++) {
                const int m = m0 + mt * 16 + e + half * 8;
                const int b = m >> 11, s = m & (SS - 1);
                const bf16 val = (bf16)(acc[mt][nt][e] + bn);
                if (z == 2) {
                    Vt[(((size_t)b * HH + hh) * DD + dd) * SS + s] = val;
                } else if (z == 0) {
                    Qo[(((size_t)b * HH + hh) * SS + s) * DD + dd] = val;
                } else {
                    Ko[(((size_t)b * HH + hh) * SS + s) * DD + dd] = val;
                }
            }
        }
    }
}

// ---------------------------------------------------------------------------
// ||q||^2 and ||k||^2 per row (from bf16 Q/K in [B,H,S,D])
// ---------------------------------------------------------------------------
__global__ void rownorm_kernel(const bf16* __restrict__ Q, const bf16* __restrict__ K,
                               float* __restrict__ qn, float* __restrict__ kn) {
    const int total = BB * HH * SS;
    int idx = blockIdx.x * blockDim.x + threadIdx.x;
    if (idx >= 2 * total) return;
    const bool isQ = idx < total;
    const int  r   = isQ ? idx : idx - total;
    const bf16* p  = (isQ ? Q : K) + (size_t)r * DD;
    float s = 0.0f;
#pragma unroll 8
    for (int d = 0; d < DD; d++) {
        float v = (float)p[d];
        s += v * v;
    }
    (isQ ? qn : kn)[r] = s;
}

// ---------------------------------------------------------------------------
// YAT flash attention. Block = 256 threads = 8 waves; each wave owns 32 Q rows.
// grid = (SS/256, B*H). Streams 32-key tiles; scores and P*V via bf16 WMMA.
// P (f32 C-layout) -> bf16 via per-wave LDS tile -> A-fragment layout.
// ---------------------------------------------------------------------------
__global__ void __launch_bounds__(256) yat_attention_kernel(
    const bf16* __restrict__ Q, const bf16* __restrict__ K, const bf16* __restrict__ Vt,
    const float* __restrict__ qn, const float* __restrict__ kn,
    const float* __restrict__ alpha, bf16* __restrict__ O)
{
    __shared__ __align__(16) bf16 Plds[8][32][40];   // 20 KB, stride 40 keeps 16B align

    const int wave = threadIdx.x >> 5;
    const int lane = threadIdx.x & 31;
    const int half = lane >> 4, lm = lane & 15;
    const int bh = blockIdx.y;
    const int b  = bh >> 4, h = bh & (HH - 1);
    const int q0 = blockIdx.x * 256 + wave * 32;

    const bf16*  Qp  = Q  + (size_t)bh * SS * DD;
    const bf16*  Kp  = K  + (size_t)bh * SS * DD;
    const bf16*  Vp  = Vt + (size_t)bh * DD * SS;
    const float* qnp = qn + (size_t)bh * SS;
    const float* knp = kn + (size_t)bh * SS;

    // (sqrt(D)/log1p(D))^alpha = (8/ln 65)^alpha
    const float scale = powf(8.0f / logf(65.0f), alpha[0]);

    // Q fragments for this wave's 32 rows (2 m-tiles x 2 k-chunks of 32)
    v16bf qf[2][2];
#pragma unroll
    for (int mt = 0; mt < 2; mt++)
#pragma unroll
        for (int kc = 0; kc < 2; kc++)
            qf[mt][kc] = load_frag_a(Qp, q0 + mt * 16 + lm, DD, kc * 32, half);

    float qn_r[2][8];
#pragma unroll
    for (int mt = 0; mt < 2; mt++)
#pragma unroll
        for (int e = 0; e < 8; e++)
            qn_r[mt][e] = qnp[q0 + mt * 16 + e + half * 8];

    float mrun[2][8], lrun[2][8];
    v8f   oacc[2][4];
#pragma unroll
    for (int mt = 0; mt < 2; mt++)
#pragma unroll
        for (int e = 0; e < 8; e++) { mrun[mt][e] = -3.0e38f; lrun[mt][e] = 0.0f; }
#pragma unroll
    for (int mt = 0; mt < 2; mt++)
#pragma unroll
        for (int dt = 0; dt < 4; dt++)
#pragma unroll
            for (int e = 0; e < 8; e++) oacc[mt][dt][e] = 0.0f;

    for (int kt = 0; kt < SS; kt += 32) {
        // ---- scores: S = Q K^T  (2x2 tiles of 16x16, k = 64 via two WMMA) ----
        v16bf kf[2][2];
#pragma unroll
        for (int nt = 0; nt < 2; nt++)
#pragma unroll
            for (int kc = 0; kc < 2; kc++)
                kf[nt][kc] = load_frag_b(Kp, kt + nt * 16 + lm, DD, kc * 32, half);

        v8f sacc[2][2];
#pragma unroll
        for (int mt = 0; mt < 2; mt++)
#pragma unroll
            for (int nt = 0; nt < 2; nt++) {
#pragma unroll
                for (int e = 0; e < 8; e++) sacc[mt][nt][e] = 0.0f;
                sacc[mt][nt] = __builtin_amdgcn_wmma_f32_16x16x32_bf16(
                    false, qf[mt][0], false, kf[nt][0], (short)0, sacc[mt][nt], false, false);
                sacc[mt][nt] = __builtin_amdgcn_wmma_f32_16x16x32_bf16(
                    false, qf[mt][1], false, kf[nt][1], (short)0, sacc[mt][nt], false, false);
            }

        float knc[2];
#pragma unroll
        for (int nt = 0; nt < 2; nt++) knc[nt] = knp[kt + nt * 16 + lm];

        // ---- YAT transform: (dot^2 / (||q||^2+||k||^2-2dot+eps)) * scale ----
#pragma unroll
        for (int mt = 0; mt < 2; mt++)
#pragma unroll
            for (int nt = 0; nt < 2; nt++)
#pragma unroll
                for (int e = 0; e < 8; e++) {
                    const float dot = sacc[mt][nt][e];
                    const float d2  = qn_r[mt][e] + knc[nt] - 2.0f * dot + EPSY;
                    sacc[mt][nt][e] = (dot * dot / d2) * scale;
                }

        // ---- online softmax (row = e + 8*half lives in a 16-lane half) ----
#pragma unroll
        for (int mt = 0; mt < 2; mt++) {
#pragma unroll
            for (int e = 0; e < 8; e++) {
                float rmax = fmaxf(sacc[mt][0][e], sacc[mt][1][e]);
#pragma unroll
                for (int off = 8; off >= 1; off >>= 1)
                    rmax = fmaxf(rmax, __shfl_xor(rmax, off, 32));
                const float mnew = fmaxf(mrun[mt][e], rmax);
                const float corr = __expf(mrun[mt][e] - mnew);
                const float p0 = __expf(sacc[mt][0][e] - mnew);
                const float p1 = __expf(sacc[mt][1][e] - mnew);
                float rs = p0 + p1;
#pragma unroll
                for (int off = 8; off >= 1; off >>= 1)
                    rs += __shfl_xor(rs, off, 32);
                lrun[mt][e] = lrun[mt][e] * corr + rs;
                mrun[mt][e] = mnew;
                sacc[mt][0][e] = p0;
                sacc[mt][1][e] = p1;
#pragma unroll
                for (int dt = 0; dt < 4; dt++) oacc[mt][dt][e] *= corr;
            }
        }

        // ---- P (C-layout f32) -> LDS bf16 [row][key] ----
#pragma unroll
        for (int mt = 0; mt < 2; mt++)
#pragma unroll
            for (int nt = 0; nt < 2; nt++)
#pragma unroll
                for (int e = 0; e < 8; e++)
                    Plds[wave][mt * 16 + e + half * 8][nt * 16 + lm] =
                        (bf16)sacc[mt][nt][e];

        // ---- read back as A-fragments (k = 32 keys) ----
        v16bf pf[2];
#pragma unroll
        for (int mt = 0; mt < 2; mt++) {
            const int row = mt * 16 + lm;
            FragBF f;
            f.h[0] = *(const v8bf*)&Plds[wave][row][half * 8];
            f.h[1] = *(const v8bf*)&Plds[wave][row][16 + half * 8];
            pf[mt] = f.v;
        }

        // ---- O += P V  (V transposed [d][s] -> contiguous B-fragments) ----
        v16bf vf[4];
#pragma unroll
        for (int dt = 0; dt < 4; dt++)
            vf[dt] = load_frag_b(Vp, dt * 16 + lm, SS, kt, half);

#pragma unroll
        for (int mt = 0; mt < 2; mt++)
#pragma unroll
            for (int dt = 0; dt < 4; dt++)
                oacc[mt][dt] = __builtin_amdgcn_wmma_f32_16x16x32_bf16(
                    false, pf[mt], false, vf[dt], (short)0, oacc[mt][dt], false, false);
    }

    // ---- epilogue: O /= l ; store bf16 into [B,S,H*D] for output GEMM ----
#pragma unroll
    for (int mt = 0; mt < 2; mt++)
#pragma unroll
        for (int e = 0; e < 8; e++) {
            const float inv = 1.0f / lrun[mt][e];
            const int s = q0 + mt * 16 + e + half * 8;
#pragma unroll
            for (int dt = 0; dt < 4; dt++) {
                const int d = dt * 16 + lm;
                O[((size_t)b * SS + s) * DMOD + h * DD + d] =
                    (bf16)(oacc[mt][dt][e] * inv);
            }
        }
}

// ---------------------------------------------------------------------------
// Output projection GEMM: f32 out = Obf16 @ woT + bo  (64x64 wave tile)
// ---------------------------------------------------------------------------
__global__ void __launch_bounds__(256) out_gemm_kernel(
    const bf16* __restrict__ A, const bf16* __restrict__ WT,
    const float* __restrict__ bias, float* __restrict__ Cout)
{
    const int wave = threadIdx.x >> 5;
    const int wm   = wave >> 2, wn = wave & 3;
    const int lane = threadIdx.x & 31;
    const int half = lane >> 4, lm = lane & 15;

    const int m0 = blockIdx.x * 128 + wm * 64;
    const int n0 = blockIdx.y * 256 + wn * 64;

    v8f acc[4][4];
#pragma unroll
    for (int mt = 0; mt < 4; mt++)
#pragma unroll
        for (int nt = 0; nt < 4; nt++)
#pragma unroll
            for (int e = 0; e < 8; e++) acc[mt][nt][e] = 0.0f;

    for (int kb = 0; kb < DMOD; kb += 32) {
        v16bf af[4], bfm[4];
#pragma unroll
        for (int mt = 0; mt < 4; mt++)
            af[mt] = load_frag_a(A, m0 + mt * 16 + lm, DMOD, kb, half);
#pragma unroll
        for (int nt = 0; nt < 4; nt++)
            bfm[nt] = load_frag_b(WT, n0 + nt * 16 + lm, DMOD, kb, half);
#pragma unroll
        for (int mt = 0; mt < 4; mt++)
#pragma unroll
            for (int nt = 0; nt < 4; nt++)
                acc[mt][nt] = __builtin_amdgcn_wmma_f32_16x16x32_bf16(
                    false, af[mt], false, bfm[nt], (short)0, acc[mt][nt], false, false);
    }

#pragma unroll
    for (int mt = 0; mt < 4; mt++)
#pragma unroll
        for (int nt = 0; nt < 4; nt++) {
            const int n  = n0 + nt * 16 + lm;
            const float bn = bias[n];
#pragma unroll
            for (int e = 0; e < 8; e++) {
                const int m = m0 + mt * 16 + e + half * 8;
                Cout[(size_t)m * DMOD + n] = acc[mt][nt][e] + bn;
            }
        }
}

// ---------------------------------------------------------------------------
// host-side launcher
// ---------------------------------------------------------------------------
extern "C" void kernel_launch(void* const* d_in, const int* in_sizes, int n_in,
                              void* d_out, int out_size, void* d_ws, size_t ws_size,
                              hipStream_t stream) {
    const float* x     = (const float*)d_in[0];
    const float* wq    = (const float*)d_in[1];
    const float* bq    = (const float*)d_in[2];
    const float* wk    = (const float*)d_in[3];
    const float* bk    = (const float*)d_in[4];
    const float* wv    = (const float*)d_in[5];
    const float* bv    = (const float*)d_in[6];
    const float* wo    = (const float*)d_in[7];
    const float* bo    = (const float*)d_in[8];
    const float* alpha = (const float*)d_in[9];

    char*  ws  = (char*)d_ws;
    size_t off = 0;
    auto alloc = [&](size_t bytes) -> void* {
        void* p = ws + off;
        off = (off + bytes + 255) & ~(size_t)255;
        return p;
    };

    const size_t nX  = (size_t)BB * SS * DMOD;    // 4M activations
    const size_t nW  = (size_t)DMOD * DMOD;       // 1M weights
    const size_t nPH = (size_t)BB * HH * SS * DD; // per-head tensor elems (== nX)

    bf16* xb   = (bf16*)alloc(nX * 2);
    bf16* wqT  = (bf16*)alloc(nW * 2);
    bf16* wkT  = (bf16*)alloc(nW * 2);
    bf16* wvT  = (bf16*)alloc(nW * 2);
    bf16* woT  = (bf16*)alloc(nW * 2);
    bf16* Qb   = (bf16*)alloc(nPH * 2);
    bf16* Kb   = (bf16*)alloc(nPH * 2);
    bf16* Vtb  = (bf16*)alloc(nPH * 2);
    float* qnb = (float*)alloc((size_t)BB * HH * SS * 4);
    float* knb = (float*)alloc((size_t)BB * HH * SS * 4);
    bf16* Ob   = (bf16*)alloc(nX * 2);
    (void)ws_size; (void)in_sizes; (void)n_in; (void)out_size;

    // 1) convert activations + transpose/convert weights to bf16
    cvt_f32_bf16_kernel<<<(int)((nX + 255) / 256), 256, 0, stream>>>(x, xb, (int)nX);
    transpose_cvt_kernel<<<(int)((nW + 255) / 256), 256, 0, stream>>>(wq, wqT);
    transpose_cvt_kernel<<<(int)((nW + 255) / 256), 256, 0, stream>>>(wk, wkT);
    transpose_cvt_kernel<<<(int)((nW + 255) / 256), 256, 0, stream>>>(wv, wvT);
    transpose_cvt_kernel<<<(int)((nW + 255) / 256), 256, 0, stream>>>(wo, woT);

    // 2) fused QKV projection (WMMA, 64x64 wave tiles)
    qkv_gemm_kernel<<<dim3(BB * SS / 128, DMOD / 256, 3), 256, 0, stream>>>(
        xb, wqT, wkT, wvT, bq, bk, bv, Qb, Kb, Vtb);

    // 3) row norms for YAT scores
    rownorm_kernel<<<(2 * BB * HH * SS + 255) / 256, 256, 0, stream>>>(Qb, Kb, qnb, knb);

    // 4) flash YAT attention (WMMA scores + WMMA P*V)
    yat_attention_kernel<<<dim3(SS / 256, BB * HH), 256, 0, stream>>>(
        Qb, Kb, Vtb, qnb, knb, alpha, Ob);

    // 5) output projection (WMMA, 64x64 wave tiles), f32 result
    out_gemm_kernel<<<dim3(BB * SS / 128, DMOD / 256), 256, 0, stream>>>(
        Ob, woT, bo, (float*)d_out);
}
